// PropAndPool_75479755260231
// MI455X (gfx1250) — matrix operand order repaired and checked by reading, not compiled, not run on hardware
//
#include <hip/hip_runtime.h>
#include <math.h>

// ---------------- problem constants (match reference) ----------------
#define GB   256          // graphs
#define NPG  500
#define NN   (GB*NPG)     // 128000 nodes
#define DEG_ 16
#define EE   (NN*DEG_)    // 2048000 edges
#define DD   128
#define KK1  400
#define KK2  320
#define KK3  256

typedef __attribute__((ext_vector_type(16))) __bf16 v16bf;
typedef __attribute__((ext_vector_type(8)))  float  v8f;

union Frag16 { uint4 q[2]; v16bf v; };

__device__ __forceinline__ unsigned short f2bf(float f) {
    union { float f; unsigned u; } c; c.f = f;
    unsigned u = c.u + 0x7FFFu + ((c.u >> 16) & 1u);   // round-to-nearest-even
    return (unsigned short)(u >> 16);
}

// ---------------- elementwise helpers ----------------
__global__ void k_fill_f32(float* __restrict__ p, float v, long long n) {
    long long i = (long long)blockIdx.x * blockDim.x + threadIdx.x;
    if (i < n) p[i] = v;
}
__global__ void k_fill_i32(int* __restrict__ p, int v, long long n) {
    long long i = (long long)blockIdx.x * blockDim.x + threadIdx.x;
    if (i < n) p[i] = v;
}
__global__ void k_cvt_bf16(unsigned short* __restrict__ dst, const float* __restrict__ src,
                           long long n) {
    long long i = (long long)blockIdx.x * blockDim.x + threadIdx.x;
    if (i < n) dst[i] = f2bf(src[i]);
}

// ---------------- weight packing into WMMA B-fragment order ----------------
// Wpack[((ntile*4+kt)*32+lane)*16+e] = bf16(W[k][n]),
//   k = kt*32 + (e>>3)*16 + (lane>>4)*8 + (e&7), n = ntile*16 + (lane&15)
__global__ void k_pack_w(const float* __restrict__ W, unsigned short* __restrict__ out) {
    int tid = blockIdx.x * blockDim.x + threadIdx.x;       // 16384 threads
    if (tid >= DD * DD) return;
    int e     = tid & 15;
    int lane  = (tid >> 4) & 31;
    int kt    = (tid >> 9) & 3;
    int ntile = tid >> 11;
    int k = kt * 32 + (e >> 3) * 16 + (lane >> 4) * 8 + (e & 7);
    int n = ntile * 16 + (lane & 15);
    out[tid] = f2bf(W[k * DD + n]);
}

// ---------------- edge scatter (mean aggregation numerator + degree) ----------------
__global__ void k_scatter(const float* __restrict__ x, const int* __restrict__ src,
                          const int* __restrict__ tgt, const float* __restrict__ valid,
                          float* __restrict__ agg, float* __restrict__ deg, long long total) {
    long long gid = (long long)blockIdx.x * blockDim.x + threadIdx.x;
    if (gid >= total) return;
    int e = (int)(gid >> 7);
    int d = (int)(gid & 127);
    float v = valid[e];
    if (v <= 0.f) return;
    int s = src[e], t = tgt[e];
    atomicAdd(&agg[(long long)t * DD + d], x[(long long)s * DD + d]);
    if (d == 0) atomicAdd(&deg[t], 1.0f);
}

__global__ void k_mean_cvt(unsigned short* __restrict__ meanb, const float* __restrict__ agg,
                           const float* __restrict__ deg, long long n) {
    long long i = (long long)blockIdx.x * blockDim.x + threadIdx.x;
    if (i >= n) return;
    float dv = fmaxf(deg[i >> 7], 1.0f);
    meanb[i] = f2bf(agg[i] / dv);
}

// ---------------- fused SAGE GEMM: H = relu(mean@Wl + x@Wr + bl) ----------------
// A tile (16 rows x 128 bf16, for both mean and x) staged into LDS ONCE per block
// via async global->LDS copies (ASYNCcnt path), then consumed by all 8 waves as
// WMMA fragments through ds_load_b128. Rows padded to 272B to rotate LDS banks.
#define ROWB 272                      // 256B of data + 16B pad, 16B aligned
__global__ __launch_bounds__(256)
void k_sage_gemm(const unsigned short* __restrict__ Ab0,   // mean, bf16 [M,128]
                 const unsigned short* __restrict__ Ab1,   // x,    bf16 [M,128]
                 const unsigned short* __restrict__ Bp0,   // Wl packed fragments
                 const unsigned short* __restrict__ Bp1,   // Wr packed fragments
                 const float* __restrict__ bias,
                 float* __restrict__ H) {
    __shared__ char smem[2 * 16 * ROWB];                   // 8704 B: [mean | x]
    int t    = threadIdx.x;
    int lane = t & 31;
    int wv   = t >> 5;        // N tile 0..7
    int m0   = blockIdx.x * 16;
    int half = lane >> 4;

    // ---- async stage: thread t copies one 16B chunk of each source tile ----
    {
        unsigned row   = (unsigned)(t >> 4);               // 0..15
        unsigned chunk = (unsigned)(t & 15);               // 0..15 (16B units)
        unsigned goff  = (unsigned)(m0 + row) * 256u + chunk * 16u;  // byte offset
        unsigned l0 = ((unsigned)(uintptr_t)(&smem[0])) + row * ROWB + chunk * 16u;
        unsigned l1 = l0 + 16u * ROWB;
        unsigned long long g0 = (unsigned long long)(uintptr_t)Ab0;
        unsigned long long g1 = (unsigned long long)(uintptr_t)Ab1;
        asm volatile("global_load_async_to_lds_b128 %0, %1, %2"
                     :: "v"(l0), "v"(goff), "s"(g0) : "memory");
        asm volatile("global_load_async_to_lds_b128 %0, %1, %2"
                     :: "v"(l1), "v"(goff), "s"(g1) : "memory");
        asm volatile("s_wait_asynccnt 0x0" ::: "memory");
    }
    __syncthreads();

    const char* base0 = smem + (lane & 15) * ROWB;         // this lane's A row (mean)
    const char* base1 = base0 + 16 * ROWB;                 // this lane's A row (x)

    v8f acc = {};
#pragma unroll
    for (int kt = 0; kt < 4; ++kt) {
        Frag16 a, b;
        a.q[0] = *(const uint4*)(base0 + (kt * 4 + half) * 16);
        a.q[1] = *(const uint4*)(base0 + (kt * 4 + 2 + half) * 16);
        const uint4* bp = (const uint4*)(Bp0 + (((size_t)wv * 4 + kt) * 32 + lane) * 16);
        b.q[0] = bp[0]; b.q[1] = bp[1];
        acc = __builtin_amdgcn_wmma_f32_16x16x32_bf16(false, a.v, false, b.v,
                                                      (short)0, acc, false, false);
    }
#pragma unroll
    for (int kt = 0; kt < 4; ++kt) {
        Frag16 a, b;
        a.q[0] = *(const uint4*)(base1 + (kt * 4 + half) * 16);
        a.q[1] = *(const uint4*)(base1 + (kt * 4 + 2 + half) * 16);
        const uint4* bp = (const uint4*)(Bp1 + (((size_t)wv * 4 + kt) * 32 + lane) * 16);
        b.q[0] = bp[0]; b.q[1] = bp[1];
        acc = __builtin_amdgcn_wmma_f32_16x16x32_bf16(false, a.v, false, b.v,
                                                      (short)0, acc, false, false);
    }
    int n = wv * 16 + (lane & 15);
    float bn = bias[n];
#pragma unroll
    for (int r = 0; r < 8; ++r) {
        int row = m0 + half * 8 + r;
        float v = acc[r] + bn;
        H[(size_t)row * DD + n] = v > 0.f ? v : 0.f;
    }
}

// ---------------- projection scores ----------------
__global__ void k_wnorm(const float* __restrict__ w, float* __restrict__ rn) {
    __shared__ float sb[128];
    int t = threadIdx.x;
    float v = w[t];
    sb[t] = v * v;
    __syncthreads();
    for (int s = 64; s; s >>= 1) { if (t < s) sb[t] += sb[t + s]; __syncthreads(); }
    if (t == 0) rn[0] = rsqrtf(sb[0]);
}

__global__ void k_score(const float* __restrict__ h, const float* __restrict__ w,
                        const float* __restrict__ rn, float* __restrict__ score, int M) {
    int lane = threadIdx.x & 31;
    int wv   = threadIdx.x >> 5;
    int row  = blockIdx.x * 8 + wv;
    if (row >= M) return;
    const float4* xr = (const float4*)(h + (size_t)row * DD);
    const float4* wr = (const float4*)w;
    float4 a = xr[lane], b = wr[lane];
    float d = a.x * b.x + a.y * b.y + a.z * b.z + a.w * b.w;
    for (int o = 16; o; o >>= 1) d += __shfl_xor(d, o);
    if (lane == 0) score[row] = d * rn[0];
}

// ---------------- top-k per graph via rank counting (matches jax top_k ordering) ----------------
__global__ void k_topk(const float* __restrict__ score, int n, int k,
                       int* __restrict__ nodemap, int* __restrict__ origin,
                       float* __restrict__ tanhv) {
    __shared__ float s[512];
    int g = blockIdx.x, t = threadIdx.x;
    for (int i = t; i < n; i += 256) s[i] = score[g * n + i];
    __syncthreads();
    for (int i = t; i < n; i += 256) {
        float si = s[i];
        int rank = 0;
        for (int j = 0; j < n; ++j) {
            float sj = s[j];
            rank += (sj > si) || (sj == si && j < i);
        }
        if (rank < k) {
            int ni = g * k + rank;
            nodemap[g * n + i] = ni;
            origin[ni] = g * n + i;
            tanhv[ni] = tanhf(si);
        }
    }
}

__global__ void k_pool_gather(float* __restrict__ xp, const float* __restrict__ h,
                              const int* __restrict__ origin, const float* __restrict__ tanhv,
                              long long count) {
    long long gid = (long long)blockIdx.x * blockDim.x + threadIdx.x;
    if (gid >= count) return;
    int j = (int)(gid >> 7), d = (int)(gid & 127);
    xp[gid] = h[(size_t)origin[j] * DD + d] * tanhv[j];
}

__global__ void k_readout(const float* __restrict__ xp, int k, float* __restrict__ z) {
    int g = blockIdx.x, d = threadIdx.x;               // 128 threads
    const float* base = xp + (size_t)g * k * DD + d;
    float mx = -INFINITY, sm = 0.f;
    for (int j = 0; j < k; ++j) {
        float v = base[(size_t)j * DD];
        mx = fmaxf(mx, v); sm += v;
    }
    z[g * 256 + d]       += mx;
    z[g * 256 + 128 + d] += sm / (float)k;
}

__global__ void k_remap(const int* __restrict__ src_in, const int* __restrict__ tgt_in,
                        const float* __restrict__ valid_in, const int* __restrict__ nodemap,
                        int* __restrict__ src_out, int* __restrict__ tgt_out,
                        float* __restrict__ valid_out, long long E) {
    long long e = (long long)blockIdx.x * blockDim.x + threadIdx.x;
    if (e >= E) return;
    float v = valid_in[e];
    int s2 = -1, t2 = -1;
    if (v > 0.f) { s2 = nodemap[src_in[e]]; t2 = nodemap[tgt_in[e]]; }
    valid_out[e] = (v > 0.f && s2 >= 0 && t2 >= 0) ? 1.f : 0.f;
    src_out[e] = s2 >= 0 ? s2 : 0;
    tgt_out[e] = t2 >= 0 ? t2 : 0;
}

// ---------------- final MLP: relu(z@W1+b1) -> relu(@W2+b2) -> sigmoid(@W3+b3) ----------------
__global__ void k_mlp(const float* __restrict__ z,
                      const float* __restrict__ W1, const float* __restrict__ b1,
                      const float* __restrict__ W2, const float* __restrict__ b2,
                      const float* __restrict__ W3, const float* __restrict__ b3,
                      float* __restrict__ out) {
    __shared__ float zz[256], z1[128], z2[64];
    int b = blockIdx.x, t = threadIdx.x;               // 128 threads
    zz[t]       = z[b * 256 + t];
    zz[t + 128] = z[b * 256 + 128 + t];
    __syncthreads();
    float acc = 0.f;
    for (int i = 0; i < 256; ++i) acc += zz[i] * W1[i * 128 + t];
    acc += b1[t];
    z1[t] = acc > 0.f ? acc : 0.f;
    __syncthreads();
    if (t < 64) {
        float a2 = 0.f;
        for (int i = 0; i < 128; ++i) a2 += z1[i] * W2[i * 64 + t];
        a2 += b2[t];
        z2[t] = a2 > 0.f ? a2 : 0.f;
    }
    __syncthreads();
    if (t == 0) {
        float a3 = 0.f;
        for (int i = 0; i < 64; ++i) a3 += z2[i] * W3[i];
        a3 += b3[0];
        out[b] = 1.f / (1.f + expf(-a3));
    }
}

// ---------------- host side ----------------
static inline unsigned blocks_for(long long n, int bs) { return (unsigned)((n + bs - 1) / bs); }

struct Scratch {
    float *valid0, *validA, *validB;
    int *srcA, *tgtA, *srcB, *tgtB;
    float *agg;              // reused as H
    float *deg;
    unsigned short *xb, *meanb, *wpack;
    float *XP0, *XP1, *score, *tanhv, *rn, *z;
    int *nodemap, *origin;
};

static void run_layer(const float* X, int M, int n, int k,
                      const int* src_in, const int* tgt_in, const float* valid_in,
                      const unsigned short* Wlp, const unsigned short* Wrp,
                      const float* bl, const float* w, const float* rn,
                      float* XPout, int* src_out, int* tgt_out, float* valid_out,
                      const Scratch& S, hipStream_t stream) {
    long long MD = (long long)M * DD;
    k_cvt_bf16<<<blocks_for(MD, 256), 256, 0, stream>>>(S.xb, X, MD);
    k_fill_f32<<<blocks_for(MD, 256), 256, 0, stream>>>(S.agg, 0.f, MD);
    k_fill_f32<<<blocks_for(M, 256), 256, 0, stream>>>(S.deg, 0.f, M);
    long long tot = (long long)EE * DD;
    k_scatter<<<blocks_for(tot, 256), 256, 0, stream>>>(X, src_in, tgt_in, valid_in,
                                                        S.agg, S.deg, tot);
    k_mean_cvt<<<blocks_for(MD, 256), 256, 0, stream>>>(S.meanb, S.agg, S.deg, MD);
    // H written into agg buffer (agg no longer needed)
    float* H = S.agg;
    k_sage_gemm<<<M / 16, 256, 0, stream>>>(S.meanb, S.xb, Wlp, Wrp, bl, H);
    k_score<<<blocks_for(M, 8), 256, 0, stream>>>(H, w, rn, S.score, M);
    k_fill_i32<<<blocks_for(M, 256), 256, 0, stream>>>(S.nodemap, -1, M);
    k_topk<<<GB, 256, 0, stream>>>(S.score, n, k, S.nodemap, S.origin, S.tanhv);
    long long cnt = (long long)GB * k * DD;
    k_pool_gather<<<blocks_for(cnt, 256), 256, 0, stream>>>(XPout, H, S.origin, S.tanhv, cnt);
    k_readout<<<GB, 128, 0, stream>>>(XPout, k, S.z);
    if (src_out)
        k_remap<<<blocks_for(EE, 256), 256, 0, stream>>>(src_in, tgt_in, valid_in, S.nodemap,
                                                         src_out, tgt_out, valid_out, EE);
}

extern "C" void kernel_launch(void* const* d_in, const int* in_sizes, int n_in,
                              void* d_out, int out_size, void* d_ws, size_t ws_size,
                              hipStream_t stream) {
    const float* x   = (const float*)d_in[0];
    const int*   src = (const int*)d_in[1];
    const int*   tgt = (const int*)d_in[2];
    const float* Wl[3] = { (const float*)d_in[3],  (const float*)d_in[7],  (const float*)d_in[11] };
    const float* bl[3] = { (const float*)d_in[4],  (const float*)d_in[8],  (const float*)d_in[12] };
    const float* Wr[3] = { (const float*)d_in[5],  (const float*)d_in[9],  (const float*)d_in[13] };
    const float* wv[3] = { (const float*)d_in[6],  (const float*)d_in[10], (const float*)d_in[14] };
    const float* W1 = (const float*)d_in[15]; const float* b1 = (const float*)d_in[16];
    const float* W2 = (const float*)d_in[17]; const float* b2 = (const float*)d_in[18];
    const float* W3 = (const float*)d_in[19]; const float* b3 = (const float*)d_in[20];

    char* p = (char*)d_ws;
    auto alloc = [&](size_t bytes) -> void* {
        void* r = (void*)p;
        p += (bytes + 255) & ~(size_t)255;
        return r;
    };
    Scratch S;
    S.valid0  = (float*)alloc((size_t)EE * 4);
    S.srcA    = (int*)  alloc((size_t)EE * 4);
    S.tgtA    = (int*)  alloc((size_t)EE * 4);
    S.validA  = (float*)alloc((size_t)EE * 4);
    S.srcB    = (int*)  alloc((size_t)EE * 4);
    S.tgtB    = (int*)  alloc((size_t)EE * 4);
    S.validB  = (float*)alloc((size_t)EE * 4);
    S.agg     = (float*)alloc((size_t)NN * DD * 4);
    S.deg     = (float*)alloc((size_t)NN * 4);
    S.xb      = (unsigned short*)alloc((size_t)NN * DD * 2);
    S.meanb   = (unsigned short*)alloc((size_t)NN * DD * 2);
    S.XP0     = (float*)alloc((size_t)GB * KK1 * DD * 4);
    S.XP1     = (float*)alloc((size_t)GB * KK1 * DD * 4);
    S.score   = (float*)alloc((size_t)NN * 4);
    S.nodemap = (int*)  alloc((size_t)NN * 4);
    S.origin  = (int*)  alloc((size_t)GB * KK1 * 4);
    S.tanhv   = (float*)alloc((size_t)GB * KK1 * 4);
    S.wpack   = (unsigned short*)alloc((size_t)6 * DD * DD * 2);
    S.rn      = (float*)alloc(64);
    S.z       = (float*)alloc((size_t)GB * 256 * 4);

    // per-call prep (deterministic: everything we read is rewritten each call)
    for (int l = 0; l < 3; ++l) {
        k_pack_w<<<64, 256, 0, stream>>>(Wl[l], S.wpack + (size_t)(2 * l)     * DD * DD);
        k_pack_w<<<64, 256, 0, stream>>>(Wr[l], S.wpack + (size_t)(2 * l + 1) * DD * DD);
        k_wnorm<<<1, 128, 0, stream>>>(wv[l], S.rn + l);
    }
    k_fill_f32<<<blocks_for((long long)GB * 256, 256), 256, 0, stream>>>(S.z, 0.f, (long long)GB * 256);
    k_fill_f32<<<blocks_for(EE, 256), 256, 0, stream>>>(S.valid0, 1.f, EE);

    // layer 1: N -> B*K1
    run_layer(x, NN, NPG, KK1, src, tgt, S.valid0,
              S.wpack + 0 * DD * DD, S.wpack + 1 * DD * DD, bl[0], wv[0], S.rn + 0,
              S.XP0, S.srcA, S.tgtA, S.validA, S, stream);
    // layer 2: B*K1 -> B*K2
    run_layer(S.XP0, GB * KK1, KK1, KK2, S.srcA, S.tgtA, S.validA,
              S.wpack + 2 * DD * DD, S.wpack + 3 * DD * DD, bl[1], wv[1], S.rn + 1,
              S.XP1, S.srcB, S.tgtB, S.validB, S, stream);
    // layer 3: B*K2 -> B*K3 (no remap needed)
    run_layer(S.XP1, GB * KK2, KK2, KK3, S.srcB, S.tgtB, S.validB,
              S.wpack + 4 * DD * DD, S.wpack + 5 * DD * DD, bl[2], wv[2], S.rn + 2,
              S.XP0, nullptr, nullptr, nullptr, S, stream);

    k_mlp<<<GB, 128, 0, stream>>>(S.z, W1, b1, W2, b2, W3, b3, (float*)d_out);
}
